// MotionVAE_19189913878712
// MI455X (gfx1250) — matrix operand order repaired and checked by reading, not compiled, not run on hardware
//
#include <hip/hip_runtime.h>
#include <hip/hip_bf16.h>
#include <stdint.h>

// ---------------- problem constants ----------------
#define B_   1024
#define T_   250
#define D_   3
#define H_   128
#define L_   32
#define G4H  512   // 4*H

typedef __attribute__((ext_vector_type(16))) _Float16 v16h;
typedef __attribute__((ext_vector_type(8)))  float    v8f;

union Frag16 {
  v16h     v;
  uint32_t u[8];
  _Float16 h[16];
};

__device__ __forceinline__ float sigmoidf_(float x) {
  return 1.0f / (1.0f + __expf(-x));
}

// CDNA5 async memory->LDS copy (ASYNCcnt). VDST = LDS byte offset, VADDR = 64b address.
__device__ __forceinline__ void async_load_b128(uint32_t lds_off, uint64_t gaddr) {
  asm volatile("global_load_async_to_lds_b128 %0, %1, off"
               :: "v"(lds_off), "v"(gaddr) : "memory");
}
__device__ __forceinline__ void wait_asynccnt(int n) {
  if (n == 0)      asm volatile("s_wait_asynccnt 0" ::: "memory");
  else if (n == 1) asm volatile("s_wait_asynccnt 1" ::: "memory");
  else             asm volatile("s_wait_asynccnt 2" ::: "memory");
}

// Build A fragment (16x32 f16) from an LDS half buffer, row stride S halfs, K offset k0.
__device__ __forceinline__ v16h load_a_frag(const _Float16* buf, int S, int k0, int lane) {
  Frag16 a;
  const int m  = lane & 15;
  const int hi = (lane >> 4) & 1;
#pragma unroll
  for (int v = 0; v < 8; ++v) {
    int koff = (hi ? 16 : 0) + 2 * v;
    a.u[v] = *(const uint32_t*)(buf + (m * S + k0 + koff));   // ds_load_b32
  }
  return a.v;
}

// Build B fragment (32x16 f16, KxN) from LDS f16 weights row-major [512][K].
__device__ __forceinline__ v16h load_b_frag(const _Float16* w, int K, int n0, int k0, int lane) {
  Frag16 b;
  const int n  = n0 + (lane & 15);
  const int kb = k0 + ((lane < 16) ? 0 : 16);
#pragma unroll
  for (int v = 0; v < 8; ++v) {
    b.u[v] = *(const uint32_t*)(w + (n * K + kb + 2 * v));    // ds_load_b32
  }
  return b.v;
}

// Same B-fragment pattern gathered once from global f32 (VGPR-resident Whh).
__device__ __forceinline__ v16h build_b_frag_f32(const float* __restrict__ w, int K,
                                                 int n0, int k0, int lane) {
  Frag16 b;
  const int n  = n0 + (lane & 15);
  const int kb = k0 + ((lane < 16) ? 0 : 16);
#pragma unroll
  for (int v = 0; v < 8; ++v) {
    b.h[2 * v]     = (_Float16)w[n * K + kb + 2 * v];
    b.h[2 * v + 1] = (_Float16)w[n * K + kb + 2 * v + 1];
  }
  return b.v;
}

// ------------------------------------------------------------------
// Persistent-time-loop LSTM layer kernel.
//   grid.x = B/16 (16 batch rows = one WMMA M tile), grid.y = directions (1|2)
//   block  = 256 threads = 8 waves; wave w owns gate columns [64w, 64w+64)
// DIN==3   : scalar input projection in gate phase (reads f32 x).
// DIN>=128 : f16 input sequence, async double-buffered LDS staging,
//            fused WMMA input projection with Wih cached f16 in LDS.
// ------------------------------------------------------------------
template<int DIN, bool DEC_SHIFT>
__global__ __launch_bounds__(256)
void lstm_kernel(const float* __restrict__ xin32, const _Float16* __restrict__ xin16,
                 int in_stride,
                 const float* __restrict__ WihF, const float* __restrict__ WhhF,
                 const float* __restrict__ bF,
                 const float* __restrict__ WihB, const float* __restrict__ WhhB,
                 const float* __restrict__ bB,
                 const float* __restrict__ h0, const float* __restrict__ c0,
                 _Float16* __restrict__ out, int out_stride,
                 float* __restrict__ htF, float* __restrict__ htB)
{
  extern __shared__ char smem_raw[];
  float*    gates  = (float*)smem_raw;                                   // [16][512] f32
  _Float16* hbuf   = (_Float16*)(smem_raw + 32768);                      // [16][128] f16
  float*    bias_s = (float*)(smem_raw + 32768 + 4096);                  // [512]  f32
  _Float16* inbuf  = (_Float16*)(smem_raw + 32768 + 4096 + 2048);        // 2x[16][DIN] f16
  char*     wih_raw = smem_raw + 32768 + 4096 + 2048 +
                      ((DIN >= 16) ? 2 * 16 * DIN * 2 : 0);
  _Float16* wih_h  = (_Float16*)wih_raw;                                 // [512][DIN] f16
  float*    wih3   = (float*)wih_raw;                                    // [512][3]  f32

  const int tid  = threadIdx.x;
  const int lane = tid & 31;
  const int wav  = tid >> 5;
  const int b0   = blockIdx.x * 16;
  const int dir  = blockIdx.y;
  const bool rev = (dir != 0);

  const float* Wih  = rev ? WihB : WihF;
  const float* Whh  = rev ? WhhB : WhhF;
  const float* bias = rev ? bB   : bF;
  float*       ht   = rev ? htB  : htF;
  const int out_off = dir * H_;

  // ---- init LDS ----
  for (int i = tid; i < G4H; i += 256) bias_s[i] = bias[i];
  if constexpr (DIN >= 16) {
    for (int i = tid; i < G4H * DIN; i += 256) wih_h[i] = (_Float16)Wih[i];
  } else {
    for (int i = tid; i < G4H * DIN; i += 256) wih3[i] = Wih[i];
  }
  for (int i = tid; i < 16 * H_; i += 256) {
    int m = i >> 7, n = i & (H_ - 1);
    hbuf[i] = h0 ? (_Float16)h0[(size_t)(b0 + m) * H_ + n] : (_Float16)0.0f;
  }

  // ---- cell state in registers: thread owns (m = tid>>4, n = (tid&15)+16k) ----
  const int cm = tid >> 4;
  const int cn = tid & 15;
  float c_reg[8];
#pragma unroll
  for (int k = 0; k < 8; ++k) {
    int n = cn + 16 * k;
    c_reg[k] = c0 ? c0[(size_t)(b0 + cm) * H_ + n] : 0.0f;
  }

  // ---- recurrent weights resident in VGPRs (B-fragments, reused every step) ----
  v16h whh[4][4];
#pragma unroll
  for (int j = 0; j < 4; ++j)
#pragma unroll
    for (int kt = 0; kt < 4; ++kt)
      whh[j][kt] = build_b_frag_f32(Whh, H_, 64 * wav + 16 * j, 32 * kt, lane);

  // ---- async staging setup (DIN>=16): 16B chunks, double-buffered ----
  constexpr int NCHUNK     = (DIN >= 16) ? 2 * DIN : 1;       // (16*DIN*2)/16
  constexpr int PER_THREAD = (DIN >= 16) ? (NCHUNK / 256) : 0; // 2 (DIN=256) / 1 (128)
  const uint32_t inbuf_lds0 = (uint32_t)(uintptr_t)(void*)inbuf; // low 32b = LDS offset

  auto stage_async = [&](int buf, int trow) {
    if constexpr (DIN >= 16) {
#pragma unroll
      for (int c = 0; c < PER_THREAD; ++c) {
        int chunk = tid + c * 256;
        int m = chunk / (DIN / 8);
        int j = chunk % (DIN / 8);
        const _Float16* g = xin16 + ((size_t)(b0 + m) * T_ + trow) * in_stride + j * 8;
        uint32_t lds = inbuf_lds0 + (uint32_t)(buf * 16 * DIN * 2 + chunk * 16);
        async_load_b128(lds, (uint64_t)(uintptr_t)g);
      }
    }
  };

  // prologue: stage step 0
  if constexpr (DIN >= 16) stage_async(0, rev ? (T_ - 1) : 0);
  __syncthreads();

  // ================= time loop =================
  for (int t = 0; t < T_; ++t) {
    const int tt = rev ? (T_ - 1 - t) : t;

    if constexpr (DIN >= 16) {
      const int nxt = t + 1;
      if (nxt < T_) {
        stage_async(nxt & 1, rev ? (T_ - 1 - nxt) : nxt);
        wait_asynccnt(PER_THREAD);   // in-order: waits for THIS step's buffer
      } else {
        wait_asynccnt(0);
      }
    }
    __syncthreads();   // all waves' async fills for buffer (t&1) have landed

    const _Float16* inb = inbuf + (t & 1) * 16 * DIN;

    // ---- per-wave GEMM: gates[16 x 64w..64w+64) = h*WhhT (+ in*WihT) ----
    v8f acc[4] = {};
#pragma unroll
    for (int kt = 0; kt < 4; ++kt) {
      v16h a = load_a_frag(hbuf, H_, 32 * kt, lane);
#pragma unroll
      for (int j = 0; j < 4; ++j)
        acc[j] = __builtin_amdgcn_wmma_f32_16x16x32_f16(
            false, a, false, whh[j][kt], (short)0, acc[j], false, false);
    }
    if constexpr (DIN >= 32) {
#pragma unroll
      for (int kt = 0; kt < DIN / 32; ++kt) {
        v16h a = load_a_frag(inb, DIN, 32 * kt, lane);
#pragma unroll
        for (int j = 0; j < 4; ++j) {
          v16h bf = load_b_frag(wih_h, DIN, 64 * wav + 16 * j, 32 * kt, lane);
          acc[j] = __builtin_amdgcn_wmma_f32_16x16x32_f16(
              false, a, false, bf, (short)0, acc[j], false, false);
        }
      }
    }

    // ---- scatter C fragments to LDS gate buffer ----
    {
      const int mhi = (lane >> 4) ? 8 : 0;
      const int nl  = lane & 15;
#pragma unroll
      for (int j = 0; j < 4; ++j) {
        int n = 64 * wav + 16 * j + nl;
#pragma unroll
        for (int r = 0; r < 8; ++r)
          gates[(r + mhi) * G4H + n] = acc[j][r];
      }
    }
    __syncthreads();

    // ---- elementwise gate combine: c,h update ----
    {
      float xv0 = 0.f, xv1 = 0.f, xv2 = 0.f;
      if constexpr (DIN == 3) {
        int srow = DEC_SHIFT ? (tt - 1) : tt;
        if (!(DEC_SHIFT && tt == 0)) {
          const float* xr = xin32 + ((size_t)(b0 + cm) * T_ + srow) * in_stride;
          xv0 = xr[0]; xv1 = xr[1]; xv2 = xr[2];
        }
      }
#pragma unroll
      for (int k = 0; k < 8; ++k) {
        int n = cn + 16 * k;
        float gi = gates[cm * G4H + n]       + bias_s[n];
        float gf = gates[cm * G4H + n + 128] + bias_s[n + 128];
        float gg = gates[cm * G4H + n + 256] + bias_s[n + 256];
        float go = gates[cm * G4H + n + 384] + bias_s[n + 384];
        if constexpr (DIN == 3) {
          gi += xv0 * wih3[(n      ) * 3 + 0] + xv1 * wih3[(n      ) * 3 + 1] + xv2 * wih3[(n      ) * 3 + 2];
          gf += xv0 * wih3[(n + 128) * 3 + 0] + xv1 * wih3[(n + 128) * 3 + 1] + xv2 * wih3[(n + 128) * 3 + 2];
          gg += xv0 * wih3[(n + 256) * 3 + 0] + xv1 * wih3[(n + 256) * 3 + 1] + xv2 * wih3[(n + 256) * 3 + 2];
          go += xv0 * wih3[(n + 384) * 3 + 0] + xv1 * wih3[(n + 384) * 3 + 1] + xv2 * wih3[(n + 384) * 3 + 2];
        }
        float iv = sigmoidf_(gi);
        float fv = sigmoidf_(gf);
        float gv = tanhf(gg);
        float ov = sigmoidf_(go);
        c_reg[k] = fv * c_reg[k] + iv * gv;
        float hv = ov * tanhf(c_reg[k]);
        hbuf[cm * H_ + n] = (_Float16)hv;
        if (out)
          out[((size_t)(b0 + cm) * T_ + tt) * out_stride + out_off + n] = (_Float16)hv;
        if (ht && t == T_ - 1)
          ht[(size_t)(b0 + cm) * H_ + n] = hv;
      }
    }
    __syncthreads();
  }
}

// ---------------- FC heads ----------------
__global__ void head1_kernel(const float* __restrict__ htf, const float* __restrict__ htb,
                             const float* __restrict__ muW, const float* __restrict__ muB,
                             const float* __restrict__ lvW, const float* __restrict__ lvB,
                             const float* __restrict__ eps,
                             float* __restrict__ mu_out, float* __restrict__ lv_out,
                             float* __restrict__ z_out)
{
  int idx = blockIdx.x * blockDim.x + threadIdx.x;   // b*32 + j
  if (idx >= B_ * L_) return;
  int b = idx >> 5, j = idx & 31;
  const float* hf = htf + (size_t)b * H_;
  const float* hb = htb + (size_t)b * H_;
  float am = muB[j], al = lvB[j];
  for (int k = 0; k < H_; ++k) {
    am += hf[k] * muW[j * 256 + k];
    al += hf[k] * lvW[j * 256 + k];
  }
  for (int k = 0; k < H_; ++k) {
    am += hb[k] * muW[j * 256 + 128 + k];
    al += hb[k] * lvW[j * 256 + 128 + k];
  }
  mu_out[idx] = am;
  lv_out[idx] = al;
  z_out[idx]  = am + eps[idx] * __expf(0.5f * al);
}

__global__ void head2_kernel(const float* __restrict__ z,
                             const float* __restrict__ hW, const float* __restrict__ hB,
                             const float* __restrict__ cW, const float* __restrict__ cB,
                             float* __restrict__ hc_h, float* __restrict__ hc_c)
{
  int idx = blockIdx.x * blockDim.x + threadIdx.x;   // b*256 + j
  if (idx >= B_ * 2 * H_) return;
  int b = idx >> 8, j = idx & 255;
  const float* zr = z + (size_t)b * L_;
  float ah = hB[j], ac = cB[j];
  for (int k = 0; k < L_; ++k) {
    ah += zr[k] * hW[j * L_ + k];
    ac += zr[k] * cW[j * L_ + k];
  }
  hc_h[idx] = ah;
  hc_c[idx] = ac;
}

__global__ __launch_bounds__(256)
void outproj_kernel(const _Float16* __restrict__ o2, const float* __restrict__ W,
                    const float* __restrict__ bo, float* __restrict__ recon)
{
  __shared__ float Ws[3 * H_];
  __shared__ float bs[3];
  for (int i = threadIdx.x; i < 3 * H_; i += 256) Ws[i] = W[i];
  if (threadIdx.x < 3) bs[threadIdx.x] = bo[threadIdx.x];
  __syncthreads();
  size_t bt = (size_t)blockIdx.x * blockDim.x + threadIdx.x;
  if (bt >= (size_t)B_ * T_) return;
  const _Float16* row = o2 + bt * H_;
  float a0 = bs[0], a1 = bs[1], a2 = bs[2];
  for (int k = 0; k < H_; ++k) {
    float v = (float)row[k];
    a0 += v * Ws[0 * H_ + k];
    a1 += v * Ws[1 * H_ + k];
    a2 += v * Ws[2 * H_ + k];
  }
  recon[bt * 3 + 0] = a0;
  recon[bt * 3 + 1] = a1;
  recon[bt * 3 + 2] = a2;
}

// ---------------- launch ----------------
static size_t lstm_smem(int din) {
  // gates 32KB + hbuf 4KB + bias 2KB + 2x inbuf + Wih
  return 32768 + 4096 + 2048 +
         ((din >= 16) ? (size_t)2 * 16 * din * 2 + (size_t)G4H * din * 2
                      : (size_t)G4H * din * 4);
}

extern "C" void kernel_launch(void* const* d_in, const int* in_sizes, int n_in,
                              void* d_out, int out_size, void* d_ws, size_t ws_size,
                              hipStream_t stream)
{
  (void)in_sizes; (void)n_in; (void)out_size; (void)ws_size;

  const float* x       = (const float*)d_in[0];
  const float* eps     = (const float*)d_in[1];
  const float* e0f_Wih = (const float*)d_in[2];
  const float* e0f_Whh = (const float*)d_in[3];
  const float* e0f_b   = (const float*)d_in[4];
  const float* e0b_Wih = (const float*)d_in[5];
  const float* e0b_Whh = (const float*)d_in[6];
  const float* e0b_b   = (const float*)d_in[7];
  const float* e1f_Wih = (const float*)d_in[8];
  const float* e1f_Whh = (const float*)d_in[9];
  const float* e1f_b   = (const float*)d_in[10];
  const float* e1b_Wih = (const float*)d_in[11];
  const float* e1b_Whh = (const float*)d_in[12];
  const float* e1b_b   = (const float*)d_in[13];
  const float* d0_Wih  = (const float*)d_in[14];
  const float* d0_Whh  = (const float*)d_in[15];
  const float* d0_b    = (const float*)d_in[16];
  const float* d1_Wih  = (const float*)d_in[17];
  const float* d1_Whh  = (const float*)d_in[18];
  const float* d1_b    = (const float*)d_in[19];
  const float* fcmuW   = (const float*)d_in[20];
  const float* fcmuB   = (const float*)d_in[21];
  const float* fclvW   = (const float*)d_in[22];
  const float* fclvB   = (const float*)d_in[23];
  const float* fchW    = (const float*)d_in[24];
  const float* fchB    = (const float*)d_in[25];
  const float* fccW    = (const float*)d_in[26];
  const float* fccB    = (const float*)d_in[27];
  const float* fcoutW  = (const float*)d_in[28];
  const float* fcoutB  = (const float*)d_in[29];

  float* outp  = (float*)d_out;
  float* recon = outp;                               // [B,T,3]
  float* mu    = outp + (size_t)B_ * T_ * 3;         // [B,32]
  float* lv    = mu + (size_t)B_ * L_;               // [B,32]

  // workspace layout: f16 sequence buffers, f32 small vectors
  char* wsb = (char*)d_ws;
  _Float16* out0 = (_Float16*)wsb;                   // [B,T,2H] f16
  size_t out0_bytes = (size_t)B_ * T_ * 2 * H_ * sizeof(_Float16);
  float* htf = (float*)(wsb + out0_bytes);           // [B,H]
  float* htb = htf + (size_t)B_ * H_;                // [B,H]
  float* z   = htb + (size_t)B_ * H_;                // [B,L]
  float* hch = z + (size_t)B_ * L_;                  // [B,2H] (layer0 | layer1 halves)
  float* hcc = hch + (size_t)B_ * 2 * H_;            // [B,2H]
  // decoder outputs alias the (dead by then) out0 region
  _Float16* o1 = out0;                               // [B,T,H] f16
  _Float16* o2 = out0 + (size_t)B_ * T_ * H_;        // [B,T,H] f16

  const dim3 blk(256);
  const dim3 grd_bi(B_ / 16, 2);   // fwd + bwd concurrently
  const dim3 grd_un(B_ / 16, 1);
  const size_t sm3 = lstm_smem(3), sm128 = lstm_smem(128), sm256 = lstm_smem(256);

  // Encoder layer 0: bidirectional, D=3 scalar input projection; writes f16 out0.
  lstm_kernel<3, false><<<grd_bi, blk, sm3, stream>>>(
      x, nullptr, D_, e0f_Wih, e0f_Whh, e0f_b, e0b_Wih, e0b_Whh, e0b_b,
      nullptr, nullptr, out0, 2 * H_, nullptr, nullptr);

  // Encoder layer 1: bidirectional, async-staged f16 input, fused 256-wide in-proj.
  lstm_kernel<256, false><<<grd_bi, blk, sm256, stream>>>(
      nullptr, out0, 2 * H_, e1f_Wih, e1f_Whh, e1f_b, e1b_Wih, e1b_Whh, e1b_b,
      nullptr, nullptr, nullptr, 0, htf, htb);

  // mu / logvar / z ; decoder init states (torch view(NL,B,H) == contiguous halves)
  head1_kernel<<<(B_ * L_ + 255) / 256, 256, 0, stream>>>(
      htf, htb, fcmuW, fcmuB, fclvW, fclvB, eps, mu, lv, z);
  head2_kernel<<<(B_ * 2 * H_ + 255) / 256, 256, 0, stream>>>(
      z, fchW, fchB, fccW, fccB, hch, hcc);

  // Decoder layer 0: D=3 shifted input (teacher forcing), init from hch/hcc halves.
  lstm_kernel<3, true><<<grd_un, blk, sm3, stream>>>(
      x, nullptr, D_, d0_Wih, d0_Whh, d0_b, d0_Wih, d0_Whh, d0_b,
      hch, hcc, o1, H_, nullptr, nullptr);

  // Decoder layer 1: async-staged f16 input, fused 128-wide in-proj.
  lstm_kernel<128, false><<<grd_un, blk, sm128, stream>>>(
      nullptr, o1, H_, d1_Wih, d1_Whh, d1_b, d1_Wih, d1_Whh, d1_b,
      hch + (size_t)B_ * H_, hcc + (size_t)B_ * H_, o2, H_, nullptr, nullptr);

  // Output projection [B,T,128] f16 -> [B,T,3] f32
  outproj_kernel<<<((B_ * T_) + 255) / 256, 256, 0, stream>>>(
      o2, fcoutW, fcoutB, recon);
}
